// DihedralBiasVmap_9113920602368
// MI455X (gfx1250) — matrix operand order repaired and checked by reading, not compiled, not run on hardware
//
#include <hip/hip_runtime.h>
#include <math.h>

typedef __attribute__((ext_vector_type(16))) _Float16 v16h;
typedef __attribute__((ext_vector_type(8)))  float    v8f;

#define N_PART   4000000
#define N_CVS    2048
#define N_MODELS 16
#define FEAT     4096   // 2*N_CVS
#define HID      128
#define LN_EPS   1e-5f
#define PI_F     3.14159265358979323846f
#define KSPLIT   16     // split-K factor for layer-0 forward GEMM

// ---- workspace byte offsets (all 64B aligned); needs ~850 KB of d_ws ----
#define OFF_FEAT   0u        // 4096 f32  raw [cos|sin]
#define OFF_DCVS   16384u    // 2048*12 f32 dphi/dx
#define OFF_XHAT   114688u   // 4096 f32  (x-mu)*invstd
#define OFF_STATS  131072u   // f32: [mu, invstd, sigma]
#define OFF_H0     131136u   // 16*128 f32 post-relu layer0
#define OFF_E      139328u   // 16 f32 per-model energy
#define OFF_GY     139392u   // 16*4096 f32 dE/d(layernorm out)
#define OFF_JAC    401536u   // 16*2048 f32 dE/dtheta
#define OFF_MF     532608u   // 2048 f32 mean forces over models
#define OFF_A0SW   540800u   // 128*32*16 f16 swizzled A (layer0 fwd)
#define OFF_GASW   671872u   // 16*4*32*16 f16 swizzled A (layer0 bwd)
#define OFF_PART   737408u   // KSPLIT*128*16 f32 split-K partials (131072 B)

// 16-bit A-fragment layout, contiguous k = 16*half + e: slot s -> K
__device__ __forceinline__ int a_slot_to_K(int s) {
  int ksp = s >> 9;            // 32-K block
  int L   = (s >> 4) & 31;     // lane
  int e   = s & 15;            // f16 element within lane
  int hh  = L >> 4;            // lane half
  int r   = e >> 1, p = e & 1;
  return (ksp << 5) + (hh << 4) + e;
}

// ------------------------------------------------------------------ K1
__global__ void dihedral_kernel(const float* __restrict__ pos,
                                const float* __restrict__ bv,
                                const int*   __restrict__ cidx,
                                float* feat_raw, float* dcvs) {
  int c = blockIdx.x * blockDim.x + threadIdx.x;
  if (c >= N_CVS) return;
  float box[3] = {bv[0], bv[4], bv[8]};
  float P[4][3];
  for (int a = 0; a < 4; ++a) {
    int p = cidx[c * 4 + a];
    for (int d = 0; d < 3; ++d) {
      float x = pos[(size_t)p * 3 + d];
      P[a][d] = x - floorf(x / box[d]) * box[d];
    }
  }
  float b1[3], b2[3], b3[3];
  for (int d = 0; d < 3; ++d) {
    b1[d] = P[1][d] - P[0][d];
    b2[d] = P[2][d] - P[1][d];
    b3[d] = P[3][d] - P[2][d];
  }
  float n1[3] = {b1[1]*b2[2]-b1[2]*b2[1], b1[2]*b2[0]-b1[0]*b2[2], b1[0]*b2[1]-b1[1]*b2[0]};
  float n2[3] = {b2[1]*b3[2]-b2[2]*b3[1], b2[2]*b3[0]-b2[0]*b3[2], b2[0]*b3[1]-b2[1]*b3[0]};
  float lb2  = sqrtf(b2[0]*b2[0]+b2[1]*b2[1]+b2[2]*b2[2]);
  float ub2[3] = {b2[0]/lb2, b2[1]/lb2, b2[2]/lb2};
  float m1[3] = {n1[1]*ub2[2]-n1[2]*ub2[1], n1[2]*ub2[0]-n1[0]*ub2[2], n1[0]*ub2[1]-n1[1]*ub2[0]};
  float x = n1[0]*n2[0]+n1[1]*n2[1]+n1[2]*n2[2];
  float y = m1[0]*n2[0]+m1[1]*n2[1]+m1[2]*n2[2];
  float phi = atan2f(y, x);
  feat_raw[c]         = cosf(phi);
  feat_raw[N_CVS + c] = sinf(phi);
  // Blondel-Karplus analytic gradient
  float n1sq = n1[0]*n1[0]+n1[1]*n1[1]+n1[2]*n1[2];
  float n2sq = n2[0]*n2[0]+n2[1]*n2[1]+n2[2]*n2[2];
  float c12 = (b1[0]*b2[0]+b1[1]*b2[1]+b1[2]*b2[2]) / (lb2*lb2);
  float c32 = (b3[0]*b2[0]+b3[1]*b2[1]+b3[2]*b2[2]) / (lb2*lb2);
  for (int d = 0; d < 3; ++d) {
    float g0 = -lb2 / n1sq * n1[d];
    float g3 =  lb2 / n2sq * n2[d];
    float g1 = (c12 - 1.f) * g0 - c32 * g3;
    float g2 = (c32 - 1.f) * g3 - c12 * g0;
    dcvs[c*12 + 0*3 + d] = g0;
    dcvs[c*12 + 1*3 + d] = g1;
    dcvs[c*12 + 2*3 + d] = g2;
    dcvs[c*12 + 3*3 + d] = g3;
  }
}

// ------------------------------------------------------------------ K2
__global__ void layernorm_kernel(const float* __restrict__ feat,
                                 const float* __restrict__ gamma,
                                 const float* __restrict__ beta,
                                 float* xhat, float* stats, _Float16* a0_sw) {
  __shared__ float r1[256], r2[256];
  int t = threadIdx.x;
  float s1 = 0.f, s2 = 0.f;
  for (int k = t; k < FEAT; k += 256) { float v = feat[k]; s1 += v; s2 += v*v; }
  r1[t] = s1; r2[t] = s2; __syncthreads();
  for (int s = 128; s > 0; s >>= 1) {
    if (t < s) { r1[t] += r1[t+s]; r2[t] += r2[t+s]; }
    __syncthreads();
  }
  float mu  = r1[0] / FEAT;
  float var = r2[0] / FEAT - mu*mu;
  float inv = rsqrtf(var + LN_EPS);
  if (t == 0) { stats[0] = mu; stats[1] = inv; }
  for (int k = t; k < FEAT; k += 256) xhat[k] = (feat[k] - mu) * inv;
  __syncthreads();
  // fill swizzled A fragments (x replicated over all 16 M rows)
  for (int s = t; s < 128 * 512; s += 256) {
    int K = a_slot_to_K(s);
    a0_sw[s] = (_Float16)(xhat[K] * gamma[K] + beta[K]);
  }
}

// ------------------------------------------------------------------ K3: layer-0 forward, models concatenated along N, split-K (WMMA)
// 2048 waves: wave = chunk(16) x tile(128); each does 8 of 128 K-steps.
__global__ void gemm0_fwd(const float* __restrict__ w0,
                          const _Float16* __restrict__ a0_sw, float* partial) {
  int wave  = (blockIdx.x * blockDim.x + threadIdx.x) >> 5;  // 0..2047
  int L     = threadIdx.x & 31;
  int tile  = wave & 127;
  int chunk = wave >> 7;           // 0..15
  int m     = tile >> 3;
  int nbase = (tile & 7) << 4;
  int hh = L >> 4, n = L & 15;
  const float* wbase = w0 + (size_t)m * FEAT * HID + nbase + n;
  v8f acc = {};
  int ks0 = chunk * (128 / KSPLIT);
  for (int ks = ks0; ks < ks0 + 128 / KSPLIT; ++ks) {
    v16h a = *(const v16h*)(a0_sw + (((ks << 5) + L) << 4));
    int Kb = (ks << 5) + (hh << 4);
    __builtin_prefetch(wbase + (size_t)(Kb + 32) * HID, 0, 1);
    v16h b;
#pragma unroll
    for (int e = 0; e < 16; ++e) b[e] = (_Float16)wbase[(size_t)(Kb + e) * HID];
    acc = __builtin_amdgcn_wmma_f32_16x16x32_f16(false, a, false, b,
                                                 (short)0, acc, false, false);
  }
  if (hh == 0)  // row M=0 lives in VGPR0 of lanes 0-15
    partial[(chunk * 128 + tile) * 16 + n] = acc[0];
}

// K3b: sum split-K partials, add bias, ReLU
__global__ void combine_h0(const float* __restrict__ partial,
                           const float* __restrict__ b0, float* h0) {
  int j = blockIdx.x * blockDim.x + threadIdx.x;   // 0..2047
  if (j >= N_CVS) return;
  int tile = j >> 4, n = j & 15;
  float s = 0.f;
#pragma unroll
  for (int c2 = 0; c2 < KSPLIT; ++c2) s += partial[(c2 * 128 + tile) * 16 + n];
  int m = tile >> 3, nloc = ((tile & 7) << 4) + n;
  h0[m * HID + nloc] = fmaxf(s + b0[m * HID + nloc], 0.f);
}

// ------------------------------------------------------------------ K4: hidden layers fwd + bwd per model (small, VALU/LDS)
__global__ void mlp_mid(const float* w1, const float* b1, const float* w2, const float* b2,
                        const float* w3, const float* b3, const float* w4, const float* b4,
                        const float* __restrict__ h0, float* E, _Float16* ga_sw) {
  __shared__ float hs[4][HID];   // post-relu activations
  __shared__ float gs[HID];
  __shared__ float red[HID];
  int m = blockIdx.x, t = threadIdx.x;
  hs[0][t] = h0[m * HID + t];
  __syncthreads();
  const float* wf[3] = {w1, w2, w3};
  const float* bf[3] = {b1, b2, b3};
  for (int l = 0; l < 3; ++l) {
    const float* w = wf[l] + (size_t)m * HID * HID;
    float acc = bf[l][m * HID + t];
    for (int k = 0; k < HID; ++k) acc += hs[l][k] * w[k * HID + t];
    hs[l + 1][t] = fmaxf(acc, 0.f);
    __syncthreads();
  }
  red[t] = hs[3][t] * w4[m * HID + t];
  __syncthreads();
  for (int s = 64; s > 0; s >>= 1) { if (t < s) red[t] += red[t + s]; __syncthreads(); }
  if (t == 0) E[m] = red[0] + b4[m];
  // reverse mode
  float g = w4[m * HID + t] * (hs[3][t] > 0.f ? 1.f : 0.f);   // g_a3
  const float* wb[3] = {w3, w2, w1};
  for (int l = 0; l < 3; ++l) {
    __syncthreads();
    gs[t] = g; __syncthreads();
    const float* w = wb[l] + (size_t)m * HID * HID;
    float acc = 0.f;
    for (int j = 0; j < HID; ++j) acc += w[t * HID + j] * gs[j];
    g = acc * (hs[2 - l][t] > 0.f ? 1.f : 0.f);
  }
  __syncthreads(); gs[t] = g; __syncthreads();   // gs = g_a0
  for (int s = t; s < 2048; s += HID) {          // swizzled A for backward GEMM (K=128)
    int K = a_slot_to_K(s);
    ga_sw[m * 2048 + s] = (_Float16)gs[K];
  }
}

// ------------------------------------------------------------------ K5: layer-0 backward per model (WMMA): gy = W0 @ g_a0
__global__ void gemm0_bwd(const float* __restrict__ w0,
                          const _Float16* __restrict__ ga_sw, float* gy) {
  int wave = (blockIdx.x * blockDim.x + threadIdx.x) >> 5;  // 0..4095
  int L = threadIdx.x & 31;
  int m    = wave >> 8;
  int tile = wave & 255;         // N-tile over 4096
  int hh = L >> 4, n = L & 15;
  int kp = (tile << 4) + n;      // output feature index
  const float* wrow = w0 + ((size_t)m * FEAT + kp) * HID;  // contiguous in j
  v8f acc = {};
#pragma unroll
  for (int ks = 0; ks < 4; ++ks) {
    v16h a = *(const v16h*)(ga_sw + m * 2048 + (((ks << 5) + L) << 4));
    int jb = (ks << 5) + (hh << 4);
    v16h b;
#pragma unroll
    for (int e = 0; e < 16; ++e) b[e] = (_Float16)wrow[jb + e];
    acc = __builtin_amdgcn_wmma_f32_16x16x32_f16(false, a, false, b,
                                                 (short)0, acc, false, false);
  }
  if (hh == 0) gy[m * FEAT + kp] = acc[0];
}

// ------------------------------------------------------------------ K6: layernorm backward + cos/sin chain -> jac
__global__ void jac_kernel(const float* __restrict__ gy, const float* __restrict__ gamma,
                           const float* __restrict__ xhat, const float* __restrict__ stats,
                           const float* __restrict__ feat_raw, float* jac) {
  __shared__ float r1[256], r2[256];
  int m = blockIdx.x, t = threadIdx.x;
  float s1 = 0.f, s2 = 0.f;
  for (int k = t; k < FEAT; k += 256) {
    float gg = gy[m * FEAT + k] * gamma[k];
    s1 += gg; s2 += gg * xhat[k];
  }
  r1[t] = s1; r2[t] = s2; __syncthreads();
  for (int s = 128; s > 0; s >>= 1) {
    if (t < s) { r1[t] += r1[t+s]; r2[t] += r2[t+s]; }
    __syncthreads();
  }
  float m1 = r1[0] / FEAT, m2 = r2[0] / FEAT, inv = stats[1];
  for (int c = t; c < N_CVS; c += 256) {
    float ggc = gy[m * FEAT + c]          * gamma[c];
    float ggs = gy[m * FEAT + N_CVS + c]  * gamma[N_CVS + c];
    float Gc = (ggc - m1 - xhat[c]         * m2) * inv;
    float Gs = (ggs - m1 - xhat[N_CVS + c] * m2) * inv;
    // d/dtheta: cos' = -sin, sin' = cos
    jac[m * N_CVS + c] = -feat_raw[N_CVS + c] * Gc + feat_raw[c] * Gs;
  }
}

// ------------------------------------------------------------------ K7: ensemble variance -> sigma, energy scalar
__global__ void reduce_kernel(const float* __restrict__ jac, const float* __restrict__ E,
                              float* mf, float* stats, float* out) {
  __shared__ float red[256];
  int t = threadIdx.x;
  float vsum = 0.f;
  for (int c = t; c < N_CVS; c += 256) {
    float mean = 0.f;
    for (int m = 0; m < N_MODELS; ++m) mean += jac[m * N_CVS + c];
    mean *= (1.f / N_MODELS);
    mf[c] = mean;
    float v = 0.f;
    for (int m = 0; m < N_MODELS; ++m) { float d = jac[m * N_CVS + c] - mean; v += d * d; }
    vsum += v * (1.f / (N_MODELS - 1));   // ddof=1
  }
  red[t] = vsum; __syncthreads();
  for (int s = 128; s > 0; s >>= 1) { if (t < s) red[t] += red[t + s]; __syncthreads(); }
  if (t == 0) {
    float md  = sqrtf(red[0] / N_CVS);
    float isw = (3.0f - md) / (3.0f - 2.0f);
    float fi  = floorf(isw);
    float sg  = fi < 0.f ? 0.f
              : (fi == 0.f ? 0.5f * (1.f + cosf(PI_F * (1.f - isw))) : 1.f);
    float Em = 0.f;
    for (int m = 0; m < N_MODELS; ++m) Em += E[m];
    out[0]   = (Em / N_MODELS) * sg;
    stats[2] = sg;
  }
}

// ------------------------------------------------------------------ K8/K9: forces
__global__ void zero_forces(float* out) {
  const size_t NF = (size_t)3 * N_PART;          // forces at out[1..NF]
  size_t i = (size_t)blockIdx.x * blockDim.x + threadIdx.x;
  size_t stride = (size_t)gridDim.x * blockDim.x;
  if (i < 3) out[1 + i] = 0.f;                   // head 1..3
  if (i == 3) out[NF] = 0.f;                     // tail
  float4* v = (float4*)(out + 4);                // out+4 is 16B aligned
  size_t nv = (NF - 3) / 4;                      // covers 4 .. NF-1
  float4 z; z.x = z.y = z.z = z.w = 0.f;
  for (size_t k = i; k < nv; k += stride) v[k] = z;
}

__global__ void scatter_forces(const int* __restrict__ cidx, const float* __restrict__ dcvs,
                               const float* __restrict__ mf, const float* __restrict__ stats,
                               float* out) {
  int c = blockIdx.x * blockDim.x + threadIdx.x;
  if (c >= N_CVS) return;
  float s = stats[2] * mf[c];
  for (int a = 0; a < 4; ++a) {
    int p = cidx[c * 4 + a];
    for (int d = 0; d < 3; ++d)
      atomicAdd(out + 1 + (size_t)p * 3 + d, s * dcvs[c * 12 + a * 3 + d]);
  }
}

// ------------------------------------------------------------------ host
extern "C" void kernel_launch(void* const* d_in, const int* in_sizes, int n_in,
                              void* d_out, int out_size, void* d_ws, size_t ws_size,
                              hipStream_t stream) {
  const float* positions = (const float*)d_in[0];
  const float* boxv      = (const float*)d_in[1];
  const int*   cidx      = (const int*)  d_in[2];
  const float* gamma     = (const float*)d_in[3];
  const float* beta      = (const float*)d_in[4];
  const float* w0 = (const float*)d_in[5],  *b0 = (const float*)d_in[6];
  const float* w1 = (const float*)d_in[7],  *b1 = (const float*)d_in[8];
  const float* w2 = (const float*)d_in[9],  *b2 = (const float*)d_in[10];
  const float* w3 = (const float*)d_in[11], *b3 = (const float*)d_in[12];
  const float* w4 = (const float*)d_in[13], *b4 = (const float*)d_in[14];
  float* out = (float*)d_out;

  char* ws = (char*)d_ws;
  float*     feat  = (float*)(ws + OFF_FEAT);
  float*     dcvs  = (float*)(ws + OFF_DCVS);
  float*     xhat  = (float*)(ws + OFF_XHAT);
  float*     stats = (float*)(ws + OFF_STATS);
  float*     h0    = (float*)(ws + OFF_H0);
  float*     E     = (float*)(ws + OFF_E);
  float*     gy    = (float*)(ws + OFF_GY);
  float*     jac   = (float*)(ws + OFF_JAC);
  float*     mf    = (float*)(ws + OFF_MF);
  _Float16*  a0_sw = (_Float16*)(ws + OFF_A0SW);
  _Float16*  ga_sw = (_Float16*)(ws + OFF_GASW);
  float*     part  = (float*)(ws + OFF_PART);

  dihedral_kernel<<<8, 256, 0, stream>>>(positions, boxv, cidx, feat, dcvs);
  layernorm_kernel<<<1, 256, 0, stream>>>(feat, gamma, beta, xhat, stats, a0_sw);
  gemm0_fwd<<<256, 256, 0, stream>>>(w0, a0_sw, part);       // 2048 waves, split-K
  combine_h0<<<8, 256, 0, stream>>>(part, b0, h0);
  mlp_mid<<<16, 128, 0, stream>>>(w1, b1, w2, b2, w3, b3, w4, b4, h0, E, ga_sw);
  gemm0_bwd<<<512, 256, 0, stream>>>(w0, ga_sw, gy);         // 4096 waves
  jac_kernel<<<16, 256, 0, stream>>>(gy, gamma, xhat, stats, feat, jac);
  reduce_kernel<<<1, 256, 0, stream>>>(jac, E, mf, stats, out);
  zero_forces<<<2048, 256, 0, stream>>>(out);
  scatter_forces<<<8, 256, 0, stream>>>(cidx, dcvs, mf, stats, out);
}
